// MLPAndGCNSerial_32298154065952
// MI455X (gfx1250) — matrix operand order repaired and checked by reading, not compiled, not run on hardware
//
#include <hip/hip_runtime.h>
#include <hip/hip_bf16.h>
#include <stdint.h>

// ---------------------------------------------------------------------------
// Types for CDNA5 WMMA (gfx1250, wave32): bf16 16x16x32 -> f32 accumulate
// ---------------------------------------------------------------------------
typedef __attribute__((ext_vector_type(16))) __bf16 v16bf;
typedef __attribute__((ext_vector_type(8)))  float  v8f;

// ---------------------------------------------------------------------------
// GEMM: Y[nrows x NCOLS] = act( X[nrows x 64] * W[64 x NCOLS] + bias )
// One wave computes a 16-row x NCOLS-col tile: NT n-tiles x 2 k-steps WMMAs.
// B fragments built once per wave (W is <=16KB, L2-resident), reused over a
// grid-stride loop across row tiles. Compile-time NCOLS/RELU/HAS_BIAS remove
// all runtime guards from the epilogue.
// ---------------------------------------------------------------------------
template <int NCOLS, bool RELU, bool HAS_BIAS>
__global__ void __launch_bounds__(256, 1)
gemm_bf16_wmma_kernel(const float* __restrict__ X,
                      const float* __restrict__ W,
                      const float* __restrict__ bias,
                      float* __restrict__ Y,
                      int nrows) {
    constexpr int NT = (NCOLS + 15) / 16;     // n-tiles (4 for 64, 3 for 40)

    const int lane = threadIdx.x & 31;
    const int hi   = lane >> 4;    // 0 | 1 (lane half)
    const int lo   = lane & 15;

    const int wavesPerBlock = blockDim.x >> 5;
    const int wid        = blockIdx.x * wavesPerBlock + (threadIdx.x >> 5);
    const int totalWaves = gridDim.x * wavesPerBlock;
    const int nTiles     = (nrows + 15) >> 4;

    // ---- Build B fragments Bf[t][s]: 32x16 bf16 slices of W ----
    // lane holds column c = t*16 + lo ; K = s*32 + hi*16 + {2v, 2v+1}
    v16bf Bf[NT][2];
#pragma unroll
    for (int t = 0; t < NT; ++t) {
        const int c = t * 16 + lo;
        const bool cin = (c < NCOLS);
#pragma unroll
        for (int s = 0; s < 2; ++s) {
#pragma unroll
            for (int v = 0; v < 8; ++v) {
                const int k0 = s * 32 + hi * 16 + 2 * v;
                const float f0 = cin ? W[k0 * NCOLS + c]       : 0.0f;
                const float f1 = cin ? W[(k0 + 1) * NCOLS + c] : 0.0f;
                Bf[t][s][2 * v]     = (__bf16)f0;
                Bf[t][s][2 * v + 1] = (__bf16)f1;
            }
        }
    }
    float bcol[NT];
#pragma unroll
    for (int t = 0; t < NT; ++t) {
        const int c = t * 16 + lo;
        bcol[t] = (HAS_BIAS && c < NCOLS) ? bias[c] : 0.0f;
    }

    for (int tile = wid; tile < nTiles; tile += totalWaves) {
        const int rowBase = tile << 4;
        int row = rowBase + lo;
        if (row >= nrows) row = nrows - 1;          // clamped load, guarded store
        const float4* xr = (const float4*)(X + (size_t)row * 64);

        // ---- A fragments Af[s]: 16x32 bf16 slice of X ----
        // lane holds row M=lo; K runs [kb, kb+16), kb = s*32 + hi*16
        v16bf Af[2];
#pragma unroll
        for (int s = 0; s < 2; ++s) {
            const int kb4 = (s * 32 + hi * 16) >> 2;   // float4 index of run start
            const float4 r0 = xr[kb4];
            const float4 r1 = xr[kb4 + 1];
            const float4 r2 = xr[kb4 + 2];
            const float4 r3 = xr[kb4 + 3];
            Af[s][0]  = (__bf16)r0.x;  Af[s][1]  = (__bf16)r0.y;
            Af[s][2]  = (__bf16)r0.z;  Af[s][3]  = (__bf16)r0.w;
            Af[s][4]  = (__bf16)r1.x;  Af[s][5]  = (__bf16)r1.y;
            Af[s][6]  = (__bf16)r1.z;  Af[s][7]  = (__bf16)r1.w;
            Af[s][8]  = (__bf16)r2.x;  Af[s][9]  = (__bf16)r2.y;
            Af[s][10] = (__bf16)r2.z;  Af[s][11] = (__bf16)r2.w;
            Af[s][12] = (__bf16)r3.x;  Af[s][13] = (__bf16)r3.y;
            Af[s][14] = (__bf16)r3.z;  Af[s][15] = (__bf16)r3.w;
        }

        // ---- 2*NT x v_wmma_f32_16x16x32_bf16 ----
        v8f acc[NT] = {};
#pragma unroll
        for (int t = 0; t < NT; ++t) {
            acc[t] = __builtin_amdgcn_wmma_f32_16x16x32_bf16(
                false, Af[0], false, Bf[t][0], (short)0, acc[t], false, false);
            acc[t] = __builtin_amdgcn_wmma_f32_16x16x32_bf16(
                false, Af[1], false, Bf[t][1], (short)0, acc[t], false, false);
        }

        // ---- Epilogue: bias (+relu) + store. C layout: (M = v + 8*hi, N = lo) ----
        const int off0 = (rowBase + 8 * hi) * NCOLS + lo;
        if (rowBase + 16 <= nrows) {               // full tile (hot path)
#pragma unroll
            for (int t = 0; t < NT; ++t) {
                if (t * 16 + lo < NCOLS) {         // compile-time true except last 40-col tile
#pragma unroll
                    for (int v = 0; v < 8; ++v) {
                        float val = acc[t][v];
                        if constexpr (HAS_BIAS) val += bcol[t];
                        if constexpr (RELU)     val = fmaxf(val, 0.0f);
                        Y[off0 + v * NCOLS + t * 16] = val;
                    }
                }
            }
        } else {                                   // row-tail tile
#pragma unroll
            for (int t = 0; t < NT; ++t) {
                if (t * 16 + lo < NCOLS) {
#pragma unroll
                    for (int v = 0; v < 8; ++v) {
                        if (rowBase + v + 8 * hi < nrows) {
                            float val = acc[t][v];
                            if constexpr (HAS_BIAS) val += bcol[t];
                            if constexpr (RELU)     val = fmaxf(val, 0.0f);
                            Y[off0 + v * NCOLS + t * 16] = val;
                        }
                    }
                }
            }
        }
    }
}

// ---------------------------------------------------------------------------
// Degree count: deg[dst[e]] += 1  (float atomics; exact for counts < 2^24)
// ---------------------------------------------------------------------------
__global__ void degree_kernel(const long long* __restrict__ dst,
                              float* __restrict__ deg, int E) {
    int e = blockIdx.x * blockDim.x + threadIdx.x;
    if (e < E) {
        int d = (int)dst[e];
        __hip_atomic_fetch_add(&deg[d], 1.0f, __ATOMIC_RELAXED,
                               __HIP_MEMORY_SCOPE_AGENT);
    }
}

// dis[i] = rsqrt(deg[i] + 1)   (in place)
__global__ void rsqrt_kernel(float* __restrict__ deg, int N) {
    int i = blockIdx.x * blockDim.x + threadIdx.x;
    if (i < N) deg[i] = rsqrtf(deg[i] + 1.0f);
}

// ---------------------------------------------------------------------------
// Edge scatter: agg[dst] += h[src] * (dis[src]*dis[dst])
// 16 lanes per edge, float4 per lane (64 channels). Gathers hit L2
// (h is 25.6MB << 192MB L2); atomics are native global_atomic_add_f32.
// ---------------------------------------------------------------------------
__global__ void scatter_kernel(const float* __restrict__ h,
                               const long long* __restrict__ src,
                               const long long* __restrict__ dst,
                               const float* __restrict__ dis,
                               float* __restrict__ agg, int E) {
    int gid = blockIdx.x * blockDim.x + threadIdx.x;
    int e   = gid >> 4;
    if (e >= E) return;
    int c4 = (gid & 15) << 2;

    int s = (int)src[e];
    int d = (int)dst[e];
    float norm = dis[s] * dis[d];

    const float4 hv = *(const float4*)(h + (size_t)s * 64 + c4);
    float* base = agg + (size_t)d * 64 + c4;
    __hip_atomic_fetch_add(base + 0, hv.x * norm, __ATOMIC_RELAXED, __HIP_MEMORY_SCOPE_AGENT);
    __hip_atomic_fetch_add(base + 1, hv.y * norm, __ATOMIC_RELAXED, __HIP_MEMORY_SCOPE_AGENT);
    __hip_atomic_fetch_add(base + 2, hv.z * norm, __ATOMIC_RELAXED, __HIP_MEMORY_SCOPE_AGENT);
    __hip_atomic_fetch_add(base + 3, hv.w * norm, __ATOMIC_RELAXED, __HIP_MEMORY_SCOPE_AGENT);
}

// ---------------------------------------------------------------------------
// Self-loop + bias (+relu):  agg[i,c] = act(agg[i,c] + h[i,c]*dis[i]^2 + b[c])
// float4 per thread (pure bandwidth pass).
// ---------------------------------------------------------------------------
__global__ void combine_kernel(float* __restrict__ agg,
                               const float* __restrict__ h,
                               const float* __restrict__ dis,
                               const float* __restrict__ bias,
                               int N, int do_relu) {
    int gid = blockIdx.x * blockDim.x + threadIdx.x;   // over N*16
    if (gid >= N * 16) return;
    int i  = gid >> 4;
    int c4 = (gid & 15) << 2;
    float di = dis[i];
    float d2 = di * di;

    const float4 a  = *(const float4*)(agg + (size_t)i * 64 + c4);
    const float4 hh = *(const float4*)(h + (size_t)i * 64 + c4);
    const float4 b  = *(const float4*)(bias + c4);
    float4 r;
    r.x = a.x + hh.x * d2 + b.x;
    r.y = a.y + hh.y * d2 + b.y;
    r.z = a.z + hh.z * d2 + b.z;
    r.w = a.w + hh.w * d2 + b.w;
    if (do_relu) {
        r.x = fmaxf(r.x, 0.0f); r.y = fmaxf(r.y, 0.0f);
        r.z = fmaxf(r.z, 0.0f); r.w = fmaxf(r.w, 0.0f);
    }
    *(float4*)(agg + (size_t)i * 64 + c4) = r;
}

// ---------------------------------------------------------------------------
// Host-side orchestration (graph-capture safe: only async ops on `stream`)
// ---------------------------------------------------------------------------
extern "C" void kernel_launch(void* const* d_in, const int* in_sizes, int n_in,
                              void* d_out, int out_size, void* d_ws, size_t ws_size,
                              hipStream_t stream) {
    const float*     x    = (const float*)d_in[0];
    const long long* eidx = (const long long*)d_in[1];   // int64 [2, E]
    const float* W1  = (const float*)d_in[2];
    const float* b1  = (const float*)d_in[3];
    const float* W2  = (const float*)d_in[4];
    const float* b2  = (const float*)d_in[5];
    const float* Wm1 = (const float*)d_in[6];
    const float* bm1 = (const float*)d_in[7];
    const float* Wm2 = (const float*)d_in[8];
    const float* bm2 = (const float*)d_in[9];
    float* out = (float*)d_out;

    const int N = in_sizes[0] / 64;
    const int E = in_sizes[1] / 2;
    const long long* src = eidx;
    const long long* dst = eidx + E;

    // Workspace layout: dis[N] | bufA[N*64] | bufB[N*64]
    char* ws = (char*)d_ws;
    float* dis = (float*)ws;
    size_t off = ((size_t)N * sizeof(float) + 255) & ~(size_t)255;
    const size_t fbytes = (((size_t)N * 64 * sizeof(float)) + 255) & ~(size_t)255;
    float* bufA = (float*)(ws + off);
    float* bufB = (float*)(ws + off + fbytes);

    const int TPB = 256;
    const int degBlocks  = (E + TPB - 1) / TPB;
    const int rsBlocks   = (N + TPB - 1) / TPB;
    const int scBlocks   = (E * 16 + TPB - 1) / TPB;
    const int cbBlocks   = (N * 16 + TPB - 1) / TPB;
    const int rowTiles   = (N + 15) / 16;
    const int gemmBlocks = (rowTiles + (TPB / 32) - 1) / (TPB / 32);

    // --- degrees: deg = segment_sum(1, dst) ; dis = rsqrt(deg + 1) ---
    hipMemsetAsync(dis, 0, (size_t)N * sizeof(float), stream);
    degree_kernel<<<degBlocks, TPB, 0, stream>>>(dst, dis, E);
    rsqrt_kernel<<<rsBlocks, TPB, 0, stream>>>(dis, N);

    // --- GCNConv 1: h = x @ W1 ; agg = scatter ; relu(agg + self + b1) ---
    gemm_bf16_wmma_kernel<64, false, false>
        <<<gemmBlocks, TPB, 0, stream>>>(x, W1, nullptr, bufA, N);
    hipMemsetAsync(bufB, 0, (size_t)N * 64 * sizeof(float), stream);
    scatter_kernel<<<scBlocks, TPB, 0, stream>>>(bufA, src, dst, dis, bufB, E);
    combine_kernel<<<cbBlocks, TPB, 0, stream>>>(bufB, bufA, dis, b1, N, 1);

    // --- GCNConv 2: h = conv1 @ W2 ; agg ; (agg + self + b2), no relu ---
    gemm_bf16_wmma_kernel<64, false, false>
        <<<gemmBlocks, TPB, 0, stream>>>(bufB, W2, nullptr, bufA, N);
    hipMemsetAsync(bufB, 0, (size_t)N * 64 * sizeof(float), stream);
    scatter_kernel<<<scBlocks, TPB, 0, stream>>>(bufA, src, dst, dis, bufB, E);
    combine_kernel<<<cbBlocks, TPB, 0, stream>>>(bufB, bufA, dis, b2, N, 0);

    // --- MLP: relu(h @ Wm1 + bm1) ; out = h @ Wm2 + bm2 ---
    gemm_bf16_wmma_kernel<64, true, true>
        <<<gemmBlocks, TPB, 0, stream>>>(bufB, Wm1, bm1, bufA, N);
    gemm_bf16_wmma_kernel<40, false, true>
        <<<gemmBlocks, TPB, 0, stream>>>(bufA, Wm2, bm2, out, N);
}
